// STC_Encoder_68556267979441
// MI455X (gfx1250) — matrix operand order repaired and checked by reading, not compiled, not run on hardware
//
#include <hip/hip_runtime.h>

// ---------------- problem constants ----------------
constexpr int B_ = 8, T_ = 24, N_ = 512, C_ = 8, D_ = 32, H_ = 64, F_ = 16;
constexpr int R_ = B_ * N_ * C_;          // 32768 rows of (b,n,c)
constexpr int L0_ = D_ + H_ + F_;          // 112
constexpr int L1_ = H_ + H_ + F_;          // 144
constexpr int Lmax_ = 144;
constexpr int Kmax_ = 6 * Lmax_;           // 864

// ---------------- types for WMMA ----------------
typedef __attribute__((ext_vector_type(16))) __bf16       v16bf;
typedef __attribute__((ext_vector_type(8)))  float        v8f;
typedef __attribute__((ext_vector_type(8)))  unsigned int v8u;

__device__ __forceinline__ unsigned short f2bf(float f) {
    unsigned int u = __float_as_uint(f);
    return (unsigned short)((u + 0x7FFFu + ((u >> 16) & 1u)) >> 16);
}
__device__ __forceinline__ float bf2f(unsigned short s) {
    return __uint_as_float(((unsigned int)s) << 16);
}

// ---------------- optional gfx1250 async global->LDS path ----------------
#if defined(__has_builtin)
#  if __has_builtin(__builtin_amdgcn_global_load_async_to_lds_b128) && \
      __has_builtin(__builtin_amdgcn_s_wait_asynccnt)
#    define ASYNC_OK 1
#  endif
#endif
#ifndef ASYNC_OK
#  define ASYNC_OK 0
#endif

#if ASYNC_OK
// param-1 type per hipcc diagnostic: 'int __attribute__((vector_size(16))) *' (generic)
typedef int v4i_vs __attribute__((__vector_size__(16)));
typedef __attribute__((address_space(3))) v4i_vs* lds_v4i_ptr;
__device__ __forceinline__ lds_v4i_ptr lds_cast(const void* p) {
    // low 32 bits of a generic LDS address are the LDS byte offset (flat aperture rules)
    return (lds_v4i_ptr)(unsigned int)(unsigned long long)p;
}
#endif

// ---------------- generic bf16 GEMM: C(f32) = A(bf16,MxK) * B(bf16,KxN) ----------------
#define BM 64
#define BN 64
#define BK 32
#define LDP 40   // ushort pitch: 80B = 20 dwords; every 8-ushort chunk 16B-aligned

__device__ __forceinline__ v16bf load_a_frag(const unsigned short* Alds, int mbase, int lane) {
    v8u u;
    int m  = mbase + (lane & 15);
    int kh = (lane < 16) ? 0 : 8;
#pragma unroll
    for (int v = 0; v < 8; ++v) {
        int k = ((v & 4) ? 16 : 0) + kh + (v & 3) * 2;   // CDNA5 16-bit A 16x32 layout
        u[v] = *(const unsigned int*)(Alds + m * LDP + k);
    }
    return __builtin_bit_cast(v16bf, u);
}
__device__ __forceinline__ v16bf load_b_frag(const unsigned short* BldsT, int cbase, int lane) {
    v8u u;
    int col = cbase + (lane & 15);
    int kh  = (lane < 16) ? 0 : 8;                       // CDNA5 16-bit B 32x16 layout
#pragma unroll
    for (int v = 0; v < 8; ++v) {
        int k = ((v & 4) ? 16 : 0) + kh + (v & 3) * 2;
        u[v] = *(const unsigned int*)(BldsT + col * LDP + k);
    }
    return __builtin_bit_cast(v16bf, u);
}

__global__ __launch_bounds__(128) void k_gemm(
    const unsigned short* __restrict__ A, int lda,
    const unsigned short* __restrict__ Bm, int ldb, long bz,
    float* __restrict__ Cm, int ldc, long cz,
    int K)
{
    __shared__ unsigned short Alds[2][BM * LDP];
    __shared__ unsigned short Blds[2][BN * LDP];   // transposed: [col][k]

    const int m0  = blockIdx.x * BM;
    const int n0  = blockIdx.y * BN;
    const unsigned short* Bz = Bm + (long)blockIdx.z * bz;
    float*                Cz = Cm + (long)blockIdx.z * cz;

    const int tid  = threadIdx.x;
    const int lane = tid & 31;
    const int wid  = tid >> 5;
    const int wy   = wid >> 1, wx = wid & 1;

    auto stage = [&](int buf, int k0) {
        // ---- A tile 64x32: 256 x 16B chunks, 2 per thread ----
#if ASYNC_OK
#pragma unroll
        for (int c2 = tid; c2 < 256; c2 += 128) {
            int r = c2 >> 2, part = c2 & 3;
            __builtin_amdgcn_global_load_async_to_lds_b128(
                (v4i_vs*)(A + (long)(m0 + r) * lda + (k0 + part * 8)),
                lds_cast(&Alds[buf][r * LDP + part * 8]),
                0, 0);
        }
#else
#pragma unroll
        for (int c2 = tid; c2 < 256; c2 += 128) {
            int r = c2 >> 2, part = c2 & 3;
            *(uint4*)(&Alds[buf][r * LDP + part * 8]) =
                *(const uint4*)(A + (long)(m0 + r) * lda + (k0 + part * 8));
        }
#endif
        // ---- B tile 32x32 (cols n0..n0+31): one b128 load per thread, transposed scatter to LDS ----
        {
            int kk = tid >> 2, q = tid & 3;
            uint4 v = *(const uint4*)(Bz + (long)(k0 + kk) * ldb + (n0 + q * 8));
            const unsigned short* s = (const unsigned short*)&v;
#pragma unroll
            for (int e = 0; e < 8; ++e)
                Blds[buf][(q * 8 + e) * LDP + kk] = s[e];
        }
    };

    v8f acc[2][2] = {};
    const int nk = K / BK;

    stage(0, 0);
    for (int kt = 0; kt < nk; ++kt) {
        const int cur = kt & 1;
#if ASYNC_OK
        __builtin_amdgcn_s_wait_asynccnt(0);
#endif
        __syncthreads();
        if (kt + 1 < nk) stage(cur ^ 1, (kt + 1) * BK);   // overlap with WMMA below

        v16bf a0 = load_a_frag(&Alds[cur][0], wy * 32,      lane);
        v16bf a1 = load_a_frag(&Alds[cur][0], wy * 32 + 16, lane);
        v16bf b0 = load_b_frag(&Blds[cur][0], wx * 32,      lane);
        v16bf b1 = load_b_frag(&Blds[cur][0], wx * 32 + 16, lane);

        acc[0][0] = __builtin_amdgcn_wmma_f32_16x16x32_bf16(false, a0, false, b0, (short)0, acc[0][0], false, false);
        acc[0][1] = __builtin_amdgcn_wmma_f32_16x16x32_bf16(false, a0, false, b1, (short)0, acc[0][1], false, false);
        acc[1][0] = __builtin_amdgcn_wmma_f32_16x16x32_bf16(false, a1, false, b0, (short)0, acc[1][0], false, false);
        acc[1][1] = __builtin_amdgcn_wmma_f32_16x16x32_bf16(false, a1, false, b1, (short)0, acc[1][1], false, false);
    }

    // epilogue: C/D layout -> VGPR r: lanes0-15 M=r, lanes16-31 M=8+r; N=lane&15
#pragma unroll
    for (int i = 0; i < 2; ++i)
#pragma unroll
        for (int j = 0; j < 2; ++j) {
            int rb = m0 + wy * 32 + i * 16 + ((lane < 16) ? 0 : 8);
            int cb = n0 + wx * 32 + j * 16 + (lane & 15);
#pragma unroll
            for (int r = 0; r < 8; ++r)
                Cz[(long)(rb + r) * ldc + cb] = acc[i][j][r];
        }
}

// ---------------- precompute: Gs^T and (2GsGs - I)^T in bf16 ----------------
__global__ void k_gs_prep(const float* __restrict__ Gs,
                          unsigned short* __restrict__ GsT,
                          unsigned short* __restrict__ Gs2T)
{
    int idx = blockIdx.x * blockDim.x + threadIdx.x;   // 512*512 threads
    int j = idx & 511;
    int i = idx >> 9;
    float s = 0.f;
    for (int k = 0; k < N_; ++k) s += Gs[j * N_ + k] * Gs[k * N_ + i];
    GsT [i * N_ + j] = f2bf(Gs[j * N_ + i]);
    Gs2T[i * N_ + j] = f2bf(2.f * s - ((i == j) ? 1.f : 0.f));
}

__global__ void k_f2bf(const float* __restrict__ src, unsigned short* __restrict__ dst, long n) {
    long i = (long)blockIdx.x * blockDim.x + threadIdx.x;
    if (i < n) dst[i] = f2bf(src[i]);
}
__global__ void k_zero(float* __restrict__ p, long n) {
    long i = (long)blockIdx.x * blockDim.x + threadIdx.x;
    if (i < n) p[i] = 0.f;
}

// ---------------- pack XH = [Xt | Ht | feature] as bf16, layout (b, n, c*L) ----------------
__global__ void k_pack_xh(const float* __restrict__ Xt, long xbstride, int Lx,
                          const float* __restrict__ Ht,
                          const float* __restrict__ feat_in,
                          unsigned short* __restrict__ XH, int L, long total)
{
    long idx = (long)blockIdx.x * blockDim.x + threadIdx.x;
    if (idx >= total) return;
    int  l  = (int)(idx % L);
    long r  = idx / L;            // (b*N + n)*C + c
    long bn = r / C_;
    int  n  = (int)(bn % N_);
    int  b  = (int)(bn / N_);
    int  c  = (int)(r % C_);
    float v;
    if (l < Lx)            v = Xt[(long)b * xbstride + ((long)n * C_ + c) * Lx + l];
    else if (l < Lx + H_)  v = Ht[r * H_ + (l - Lx)];
    else                   v = feat_in[(long)n * F_ + (l - Lx - H_)];
    XH[idx] = f2bf(v);            // XH index == idx by construction
}

// ---------------- build feat (32768 x 6L) bf16 from XH(s=0) + Y(s=1,2) + Gc mix ----------------
__global__ void k_feat(const unsigned short* __restrict__ XH,
                       const float* __restrict__ Y, long sstride,
                       const float* __restrict__ Gc,
                       unsigned short* __restrict__ feat,
                       int L, int K, long total)
{
    long idx = (long)blockIdx.x * blockDim.x + threadIdx.x;
    if (idx >= total) return;
    int  l   = (int)(idx % L);
    long row = idx / L;           // (b*N+m)*C + d
    int  d   = (int)(row % C_);
    long bn  = row / C_;
    long base = bn * (long)(C_ * L) + l;
    float src[3][C_];
#pragma unroll
    for (int c = 0; c < C_; ++c) {
        src[0][c] = bf2f(XH[base + (long)c * L]);
        src[1][c] = Y[base + (long)c * L];
        src[2][c] = Y[sstride + base + (long)c * L];
    }
    unsigned short* fo = feat + row * (long)K + l;
#pragma unroll
    for (int s = 0; s < 3; ++s) {
        float t1 = 0.f;
#pragma unroll
        for (int c = 0; c < C_; ++c) t1 += Gc[c * C_ + d] * src[s][c];
        fo[(s * 2 + 0) * L] = f2bf(src[s][d]);   // t=0: Gc_set[0]=I
        fo[(s * 2 + 1) * L] = f2bf(t1);          // t=1: Gc mix
    }
}

// ---------------- gates: u=sig, r=sig; write U, patch XH H-block with r*Ht ----------------
__global__ void k_gate(const float* __restrict__ g, const float* __restrict__ bg,
                       const float* __restrict__ Ht,
                       float* __restrict__ U,
                       unsigned short* __restrict__ XH, int Lx, int L, long total)
{
    long idx = (long)blockIdx.x * blockDim.x + threadIdx.x;
    if (idx >= total) return;
    int  h   = (int)(idx % H_);
    long row = idx / H_;
    float u  = g[row * 128 + h] + bg[h];
    float rr = g[row * 128 + 64 + h] + bg[64 + h];
    u  = 1.f / (1.f + __expf(-u));
    rr = 1.f / (1.f + __expf(-rr));
    U[idx] = u;
    float hr = rr * Ht[idx];
    int  c  = (int)(row % C_);
    long bn = row / C_;
    XH[bn * (long)(C_ * L) + (long)c * L + Lx + h] = f2bf(hr);
}

// ---------------- H update + write outputs ----------------
__global__ void k_hstep(const float* __restrict__ Cpre, const float* __restrict__ bc,
                        const float* __restrict__ U,
                        float* __restrict__ Ht,
                        float* __restrict__ out,
                        int layer, int t, long total)
{
    long idx = (long)blockIdx.x * blockDim.x + threadIdx.x;
    if (idx >= total) return;
    int  h   = (int)(idx % H_);
    long row = idx / H_;
    float cd = tanhf(Cpre[idx] + bc[h]);
    float u  = U[idx];
    float hn = (1.f - u) * Ht[idx] + u * cd;
    Ht[idx] = hn;
    int  c  = (int)(row % C_);
    long bn = row / C_;
    int  n  = (int)(bn % N_);
    int  b  = (int)(bn / N_);
    long o1 = ((((long)layer * B_ + b) * T_ + t) * N_ + n) * (C_ * H_) + (long)c * H_ + h;
    out[o1] = hn;
    if (t == T_ - 1) {
        long o2 = (long)2 * B_ * T_ * N_ * C_ * H_ +
                  (((long)layer * B_ + b) * N_ + n) * (C_ * H_) + (long)c * H_ + h;
        out[o2] = hn;
    }
}

// ---------------- host orchestration ----------------
extern "C" void kernel_launch(void* const* d_in, const int* in_sizes, int n_in,
                              void* d_out, int out_size, void* d_ws, size_t ws_size,
                              hipStream_t stream)
{
    (void)in_sizes; (void)n_in; (void)out_size; (void)ws_size;
    const float* Gs      = (const float*)d_in[0];
    const float* Gc      = (const float*)d_in[1];
    const float* X_seq   = (const float*)d_in[2];
    const float* feature = (const float*)d_in[3];
    const float* Wg[2]   = { (const float*)d_in[4], (const float*)d_in[8]  };
    const float* bg[2]   = { (const float*)d_in[5], (const float*)d_in[9]  };
    const float* Wc[2]   = { (const float*)d_in[6], (const float*)d_in[10] };
    const float* bc[2]   = { (const float*)d_in[7], (const float*)d_in[11] };
    float* out = (float*)d_out;

    char* ws = (char*)d_ws;
    size_t off = 0;
    auto carve = [&](size_t bytes) -> void* {
        void* p = ws + off;
        off = (off + bytes + 255) & ~(size_t)255;
        return p;
    };

    unsigned short* GsT   = (unsigned short*)carve((size_t)N_ * N_ * 2);
    unsigned short* Gs2T  = (unsigned short*)carve((size_t)N_ * N_ * 2);
    unsigned short* Wgb[2] = {
        (unsigned short*)carve((size_t)(6 * L0_) * 128 * 2),
        (unsigned short*)carve((size_t)(6 * L1_) * 128 * 2) };
    unsigned short* Wcb[2] = {
        (unsigned short*)carve((size_t)(6 * L0_) * 64 * 2),
        (unsigned short*)carve((size_t)(6 * L1_) * 64 * 2) };
    unsigned short* XH   = (unsigned short*)carve((size_t)R_ * Lmax_ * 2);
    float*          Yf   = (float*)carve((size_t)2 * R_ * Lmax_ * 4);
    unsigned short* feat = (unsigned short*)carve((size_t)R_ * Kmax_ * 2);
    float*          gbuf = (float*)carve((size_t)R_ * 128 * 4);   // also reused as Cpre
    float*          Ubuf = (float*)carve((size_t)R_ * H_ * 4);
    float*          Ht   = (float*)carve((size_t)R_ * H_ * 4);

    // ---- precompute ----
    k_gs_prep<<<(N_ * N_) / 256, 256, 0, stream>>>(Gs, GsT, Gs2T);
    {
        long n;
        n = (long)(6 * L0_) * 128; k_f2bf<<<(n + 255) / 256, 256, 0, stream>>>(Wg[0], Wgb[0], n);
        n = (long)(6 * L0_) * 64;  k_f2bf<<<(n + 255) / 256, 256, 0, stream>>>(Wc[0], Wcb[0], n);
        n = (long)(6 * L1_) * 128; k_f2bf<<<(n + 255) / 256, 256, 0, stream>>>(Wg[1], Wgb[1], n);
        n = (long)(6 * L1_) * 64;  k_f2bf<<<(n + 255) / 256, 256, 0, stream>>>(Wc[1], Wcb[1], n);
    }

    for (int layer = 0; layer < 2; ++layer) {
        const int  L  = (layer == 0) ? L0_ : L1_;
        const int  Lx = (layer == 0) ? D_  : H_;
        const int  K6 = 6 * L;
        const int  CL = C_ * L;
        const long sstride = (long)B_ * N_ * CL;   // Y per-s stride

        { long n = (long)R_ * H_;
          k_zero<<<(n + 255) / 256, 256, 0, stream>>>(Ht, n); }

        for (int t = 0; t < T_; ++t) {
            const float* Xbase    = (layer == 0) ? (X_seq + (long)t * N_ * C_ * D_)
                                                 : (out   + (long)t * N_ * C_ * H_);
            const long   xbstride = (layer == 0) ? (long)T_ * N_ * C_ * D_
                                                 : (long)T_ * N_ * C_ * H_;

            long totXH = (long)R_ * L;
            k_pack_xh<<<(totXH + 255) / 256, 256, 0, stream>>>(
                Xbase, xbstride, Lx, Ht, feature, XH, L, totXH);

            auto spatial = [&]() {
                dim3 grid(N_ / BM, CL / BN, B_);
                k_gemm<<<grid, 128, 0, stream>>>(GsT,  N_, XH, CL, (long)N_ * CL,
                                                 Yf,            CL, (long)N_ * CL, N_);
                k_gemm<<<grid, 128, 0, stream>>>(Gs2T, N_, XH, CL, (long)N_ * CL,
                                                 Yf + sstride,  CL, (long)N_ * CL, N_);
            };

            // ---- gate path ----
            spatial();
            k_feat<<<(totXH + 255) / 256, 256, 0, stream>>>(XH, Yf, sstride, Gc, feat, L, K6, totXH);
            k_gemm<<<dim3(R_ / BM, 128 / BN, 1), 128, 0, stream>>>(
                feat, K6, Wgb[layer], 128, 0, gbuf, 128, 0, K6);
            { long n = (long)R_ * H_;
              k_gate<<<(n + 255) / 256, 256, 0, stream>>>(gbuf, bg[layer], Ht, Ubuf, XH, Lx, L, n); }

            // ---- candidate path (XH H-block now holds reset*Ht) ----
            spatial();
            k_feat<<<(totXH + 255) / 256, 256, 0, stream>>>(XH, Yf, sstride, Gc, feat, L, K6, totXH);
            k_gemm<<<dim3(R_ / BM, 64 / BN, 1), 128, 0, stream>>>(
                feat, K6, Wcb[layer], 64, 0, gbuf /*Cpre*/, 64, 0, K6);

            { long n = (long)R_ * H_;
              k_hstep<<<(n + 255) / 256, 256, 0, stream>>>(gbuf, bc[layer], Ubuf, Ht, out, layer, t, n); }
        }
    }
}
